// CausalHAttention1D_37804302139519
// MI455X (gfx1250) — compile-verified, matrix operand order checked by reading, not run on hardware
//
#include <hip/hip_runtime.h>
#include <hip/hip_bf16.h>
#include <cstddef>
#include <cstdint>

// ---------------------------------------------------------------------------
// CausalHAttention1D for MI455X (gfx1250, wave32, WMMA + TDM).
// b=4, n=4096, dim=1024, heads=16, dim_head=64, block=16.
// Matrix math on v_wmma_f32_16x16x32_f16 (f16 in, f32 accumulate).
// Output GEMM stages its f16 A tiles through the Tensor Data Mover
// (tensor_load_to_lds + s_wait_tensorcnt), with D# LDS padding producing the
// 40-half row stride the fragment loads expect.
// Workspace requirement: ~302 MB.
// ---------------------------------------------------------------------------

#define BATCH   4
#define SEQ     4096
#define DIM     1024
#define HEADS   16
#define DIMH    64
#define BH      (BATCH * HEADS)   // 64
#define INNER   (HEADS * DIMH)    // 1024
#define NQKV    (3 * INNER)       // 3072
#define PYRLEN  4064              // sum of 4096>>l for l=1..7, per bh

typedef __attribute__((ext_vector_type(16))) _Float16 v16h;
typedef __attribute__((ext_vector_type(8)))  _Float16 v8h;
typedef __attribute__((ext_vector_type(8)))  float    v8f;
typedef __attribute__((ext_vector_type(4)))  unsigned int u32x4;
typedef __attribute__((ext_vector_type(8)))  int          i32x8;
typedef __attribute__((ext_vector_type(4)))  int          i32x4;

__device__ inline v8f v8f_zero() {
  v8f z;
#pragma unroll
  for (int i = 0; i < 8; ++i) z[i] = 0.f;
  return z;
}

// Load one 16x32 (rows x K) f16 fragment from a row-major [16][ldk] array.
// CDNA5 16-bit A layout: lane L (m = L%16, hi = L/16) holds
// K = {hi*8 .. hi*8+7} in elements 0..7 and K = {16+hi*8 ..} in elements 8..15.
// B (32x16, KxN) uses the symmetric layout with lane = n.
__device__ inline v16h frag_load(const _Float16* base, int ldk) {
  const int lane = threadIdx.x & 31;
  const int r = lane & 15, hi = lane >> 4;
  v8h lo = *(const v8h*)(base + r * ldk + hi * 8);
  v8h hf = *(const v8h*)(base + r * ldk + 16 + hi * 8);
  v16h f;
#pragma unroll
  for (int i = 0; i < 8; ++i) { f[i] = lo[i]; f[8 + i] = hf[i]; }
  return f;
}

__device__ inline v8f wmma_f16(v16h a, v16h b, v8f c) {
  // (neg_a, A, neg_b, B, c_mod, C, reuse_a, reuse_b)
  return __builtin_amdgcn_wmma_f32_16x16x32_f16(false, a, false, b, (short)0, c,
                                                false, false);
}

// ---------------------------------------------------------------------------
// TDM: DMA a 2-D f16 tile (rows x row_elems halves, global row stride
// src_stride_elems) into LDS at lds_off. D# padding: after every 16 DWORDs
// (= 32 halves = one tile row) insert 4 DWORDs (8 halves) -> LDS row stride
// 40 halves, matching frag_load's bank-conflict padding.
// ---------------------------------------------------------------------------
__device__ inline void tdm_load_tile_f16(const _Float16* gsrc,
                                         unsigned lds_off, int rows,
                                         int row_elems, int src_stride_elems) {
  unsigned long long ga = (unsigned long long)(uintptr_t)gsrc;
  u32x4 g0;
  g0[0] = 1u;                                   // count=1, user descriptor
  g0[1] = lds_off;                              // lds_addr (bytes)
  g0[2] = (unsigned)(ga & 0xffffffffu);         // global_addr[31:0]
  g0[3] = (unsigned)((ga >> 32) & 0x01ffffffu)  // global_addr[56:32]
          | (2u << 30);                         // type = 2 ("image")
  const unsigned td0 = 1u << 20, td1 = 1u << 20;  // huge dims: no OOB clip
  i32x8 g1;
  g1[0] = (1 << 16)                             // data_size = 2 bytes
        | (1 << 20)                             // pad_enable
        | (3 << 22)                             // pad_interval = 16 DWORDs
        | (3 << 25);                            // pad_amount   = 4 DWORDs
  g1[1] = (int)((td0 & 0xffffu) << 16);         // atomic_addr=0 | td0[15:0]
  g1[2] = (int)(((td0 >> 16) & 0xffffu) | ((td1 & 0xffffu) << 16));
  g1[3] = (int)(((td1 >> 16) & 0xffffu) | ((unsigned)row_elems << 16));
  g1[4] = rows & 0xffff;                        // tile_dim1 | tile_dim2=0
  g1[5] = src_stride_elems;                     // tensor_dim0_stride[31:0]
  g1[6] = 0;
  g1[7] = 0;
  i32x4 z4;
  z4[0] = 0; z4[1] = 0; z4[2] = 0; z4[3] = 0;
#if __clang_major__ >= 23
  i32x8 z8;
#pragma unroll
  for (int i = 0; i < 8; ++i) z8[i] = 0;
  __builtin_amdgcn_tensor_load_to_lds(g0, g1, z4, z4, z8, 0);
#else
  __builtin_amdgcn_tensor_load_to_lds(g0, g1, z4, z4, 0);
#endif
}

// ---------------------------------------------------------------------------
// Kernel 1: qkv = x @ w_qkv^T, epilogue scatters f16 into head-split Q0/K0/V0
// (q scaled by dim_head^-0.5 = 0.125). Tile 256(M) x 64(N), 8 waves; each
// wave owns 32 rows -> 2 A-frags x 4 B-frags = 8 WMMA per K-step of 32.
// ---------------------------------------------------------------------------
__global__ __launch_bounds__(256) void qkv_gemm_kernel(
    const float* __restrict__ X, const float* __restrict__ W,
    _Float16* __restrict__ Q0, _Float16* __restrict__ K0,
    _Float16* __restrict__ V0) {
  __shared__ _Float16 As[256 * 40];
  __shared__ _Float16 Bs[64 * 40];
  const int t = threadIdx.x, w = t >> 5, lane = t & 31;
  const int hi = lane >> 4, nn = lane & 15;
  const int n0 = blockIdx.x * 64, m0 = blockIdx.y * 256;

  v8f acc[2][4];
#pragma unroll
  for (int s = 0; s < 2; ++s)
#pragma unroll
    for (int j = 0; j < 4; ++j) acc[s][j] = v8f_zero();

  for (int k0 = 0; k0 < DIM; k0 += 32) {
#pragma unroll
    for (int i = 0; i < 8; ++i) {            // A: 256x32 f32 -> f16 LDS
      int idx = t + 256 * i;                 // 0..2047
      int row = idx >> 3, c = (idx & 7) * 4;
      float4 v = *(const float4*)(X + (size_t)(m0 + row) * DIM + k0 + c);
      _Float16* d = &As[row * 40 + c];
      d[0] = (_Float16)v.x; d[1] = (_Float16)v.y;
      d[2] = (_Float16)v.z; d[3] = (_Float16)v.w;
    }
#pragma unroll
    for (int i = 0; i < 2; ++i) {            // B: 64x32 f32 -> f16 LDS
      int idx = t + 256 * i;                 // 0..511
      int row = idx >> 3, c = (idx & 7) * 4;
      float4 v = *(const float4*)(W + (size_t)(n0 + row) * DIM + k0 + c);
      _Float16* d = &Bs[row * 40 + c];
      d[0] = (_Float16)v.x; d[1] = (_Float16)v.y;
      d[2] = (_Float16)v.z; d[3] = (_Float16)v.w;
    }
    if (k0 + 32 < DIM) {                     // global_prefetch of next slab
      __builtin_prefetch(X + (size_t)(m0 + (t >> 3)) * DIM + k0 + 32, 0, 0);
      __builtin_prefetch(W + (size_t)(n0 + (t & 63)) * DIM + k0 + 32, 0, 0);
    }
    __syncthreads();
    v16h a0 = frag_load(&As[(w * 32) * 40], 40);
    v16h a1 = frag_load(&As[(w * 32 + 16) * 40], 40);
#pragma unroll
    for (int j = 0; j < 4; ++j) {
      v16h b = frag_load(&Bs[j * 16 * 40], 40);
      acc[0][j] = wmma_f16(a0, b, acc[0][j]);
      acc[1][j] = wmma_f16(a1, b, acc[1][j]);
    }
    __syncthreads();
  }

  // Epilogue: scatter into Q/K/V, layout [bh][p][d].
#pragma unroll
  for (int s = 0; s < 2; ++s) {
#pragma unroll
    for (int j = 0; j < 4; ++j) {
      int nq = n0 + j * 16 + nn;
      int which = nq >> 10;                  // 0=q 1=k 2=v
      int rem = nq & 1023;
      int h = rem >> 6, d = rem & 63;
      _Float16* dst = (which == 0) ? Q0 : (which == 1) ? K0 : V0;
      float sc = (which == 0) ? 0.125f : 1.0f;
#pragma unroll
      for (int r = 0; r < 8; ++r) {
        int m = m0 + w * 32 + s * 16 + r + 8 * hi;
        int b = m >> 12, p = m & 4095;
        int bh = b * HEADS + h;
        dst[((size_t)bh * SEQ + p) * DIMH + d] = (_Float16)(acc[s][j][r] * sc);
      }
    }
  }
}

// ---------------------------------------------------------------------------
// Kernel 2: pyramid coarsening (q,k mean of pairs; v sum of pairs), f16.
// ---------------------------------------------------------------------------
__global__ __launch_bounds__(256) void pyr_kernel(
    const _Float16* __restrict__ Qs, const _Float16* __restrict__ Ks,
    const _Float16* __restrict__ Vs, _Float16* __restrict__ Qd,
    _Float16* __restrict__ Kd, _Float16* __restrict__ Vd, int Ld) {
  size_t total = (size_t)BH * Ld * DIMH;
  size_t gid = (size_t)blockIdx.x * blockDim.x + threadIdx.x;
  if (gid >= total) return;
  int d = (int)(gid % DIMH);
  size_t rem = gid / DIMH;
  int pos = (int)(rem % Ld);
  int bh = (int)(rem / Ld);
  size_t s0 = (((size_t)bh * 2 * Ld) + 2 * pos) * DIMH + d;
  size_t s1 = s0 + DIMH;
  Qd[gid] = (_Float16)(0.5f * ((float)Qs[s0] + (float)Qs[s1]));
  Kd[gid] = (_Float16)(0.5f * ((float)Ks[s0] + (float)Ks[s1]));
  Vd[gid] = (_Float16)((float)Vs[s0] + (float)Vs[s1]);
}

// ---------------------------------------------------------------------------
// Kernel 3: level-0 diagonal causal 16x16 block attention. 1 wave per block.
// S = qk^T (2 WMMAs, K=64), masked row exp in LDS, Y = A @ V (4 WMMAs with
// V transposed + zero-padded in LDS). Writes Ysum/Asum (=).
// ---------------------------------------------------------------------------
__global__ __launch_bounds__(32) void attn_l0_kernel(
    const _Float16* __restrict__ Q0, const _Float16* __restrict__ K0,
    const _Float16* __restrict__ V0, float* __restrict__ Ysum,
    float* __restrict__ Asum) {
  const int blk = blockIdx.x, bh = blockIdx.y;
  const int lane = threadIdx.x, hi = lane >> 4, nn = lane & 15;
  const size_t rowbase = ((size_t)bh * SEQ + blk * 16) * DIMH;
  const _Float16* q = Q0 + rowbase;
  const _Float16* k = K0 + rowbase;
  const _Float16* v = V0 + rowbase;

  __shared__ _Float16 Vt[64 * 40];   // Vt[d][kk]; kk 16..31 zero-padded
  __shared__ _Float16 Aex[16 * 40];  // exp weights; K cols 16..31 zero
  __shared__ float Sf[16 * 17];

  for (int idx = lane; idx < 16 * DIMH; idx += 32) {
    int r = idx >> 6, d = idx & 63;
    Vt[d * 40 + r] = v[idx];
  }
  for (int idx = lane; idx < 64 * 16; idx += 32) {
    int d = idx >> 4, kk = 16 + (idx & 15);
    Vt[d * 40 + kk] = (_Float16)0.f;
  }
  for (int idx = lane; idx < 16 * 16; idx += 32) {
    int r = idx >> 4, c = 16 + (idx & 15);
    Aex[r * 40 + c] = (_Float16)0.f;
  }
  __syncthreads();

  v8f s = v8f_zero();
  s = wmma_f16(frag_load(q, DIMH), frag_load(k, DIMH), s);
  s = wmma_f16(frag_load(q + 32, DIMH), frag_load(k + 32, DIMH), s);
#pragma unroll
  for (int r = 0; r < 8; ++r) Sf[(r + 8 * hi) * 17 + nn] = s[r];
  __syncthreads();

  if (lane < 16) {
    int row = lane;
    float mx = -3.0e38f;
    for (int j = 0; j <= row; ++j) mx = fmaxf(mx, Sf[row * 17 + j]);
    float sum = 0.f;
    for (int j = 0; j < 16; ++j) {
      float e = (j <= row) ? __expf(Sf[row * 17 + j] - mx) : 0.f;
      Aex[row * 40 + j] = (_Float16)e;
      sum += e;
    }
    Asum[(size_t)bh * SEQ + blk * 16 + row] = sum;
  }
  __syncthreads();

  v16h aA = frag_load(Aex, 40);
#pragma unroll
  for (int j = 0; j < 4; ++j) {
    v16h bV = frag_load(&Vt[j * 16 * 40], 40);
    v8f y = wmma_f16(aA, bV, v8f_zero());
#pragma unroll
    for (int r = 0; r < 8; ++r)
      Ysum[((size_t)bh * SEQ + blk * 16 + r + 8 * hi) * DIMH + j * 16 + nn] =
          y[r];
  }
}

// ---------------------------------------------------------------------------
// Kernel 4: level l>=1 off-diagonal block attention on the coarsened pyramid.
// Second block of each pair attends to the first. The reference keep-mask
// reduces to p mod 2^l == 2^l-1, so each coarse row contributes to exactly
// one output position: p = ((cp+1)<<l)-1. Plain += (one writer per p per
// level; levels are separate ordered launches -> deterministic).
// ---------------------------------------------------------------------------
__global__ __launch_bounds__(32) void attn_off_kernel(
    const _Float16* __restrict__ Pq, const _Float16* __restrict__ Pk,
    const _Float16* __restrict__ Pv, float* __restrict__ Ysum,
    float* __restrict__ Asum, int l) {
  const int i = blockIdx.x, bh = blockIdx.y;
  const int lane = threadIdx.x, hi = lane >> 4, nn = lane & 15;
  const int Ll = SEQ >> l;
  const size_t off = (size_t)BH * DIMH * (size_t)(SEQ - (8192 >> l));
  const _Float16* q = Pq + off + ((size_t)bh * Ll + (2 * i + 1) * 16) * DIMH;
  const _Float16* k = Pk + off + ((size_t)bh * Ll + (2 * i) * 16) * DIMH;
  const _Float16* v = Pv + off + ((size_t)bh * Ll + (2 * i) * 16) * DIMH;

  __shared__ _Float16 Vt[64 * 40];
  __shared__ _Float16 Aex[16 * 40];
  __shared__ float Sf[16 * 17];

  for (int idx = lane; idx < 16 * DIMH; idx += 32) {
    int r = idx >> 6, d = idx & 63;
    Vt[d * 40 + r] = v[idx];
  }
  for (int idx = lane; idx < 64 * 16; idx += 32) {
    int d = idx >> 4, kk = 16 + (idx & 15);
    Vt[d * 40 + kk] = (_Float16)0.f;
  }
  for (int idx = lane; idx < 16 * 16; idx += 32) {
    int r = idx >> 4, c = 16 + (idx & 15);
    Aex[r * 40 + c] = (_Float16)0.f;
  }
  __syncthreads();

  v8f s = v8f_zero();
  s = wmma_f16(frag_load(q, DIMH), frag_load(k, DIMH), s);
  s = wmma_f16(frag_load(q + 32, DIMH), frag_load(k + 32, DIMH), s);
#pragma unroll
  for (int r = 0; r < 8; ++r) Sf[(r + 8 * hi) * 17 + nn] = s[r];
  __syncthreads();

  if (lane < 16) {
    int row = lane;
    float mx = -3.0e38f;
    for (int j = 0; j < 16; ++j) mx = fmaxf(mx, Sf[row * 17 + j]);
    float sum = 0.f;
    for (int j = 0; j < 16; ++j) {
      float e = __expf(Sf[row * 17 + j] - mx);
      Aex[row * 40 + j] = (_Float16)e;
      sum += e;
    }
    int cp = (2 * i + 1) * 16 + row;
    int p = ((cp + 1) << l) - 1;
    Asum[(size_t)bh * SEQ + p] += sum;
  }
  __syncthreads();

  v16h aA = frag_load(Aex, 40);
#pragma unroll
  for (int j = 0; j < 4; ++j) {
    v16h bV = frag_load(&Vt[j * 16 * 40], 40);
    v8f y = wmma_f16(aA, bV, v8f_zero());
#pragma unroll
    for (int r = 0; r < 8; ++r) {
      int cp = (2 * i + 1) * 16 + r + 8 * hi;
      int p = ((cp + 1) << l) - 1;
      Ysum[((size_t)bh * SEQ + p) * DIMH + j * 16 + nn] += y[r];
    }
  }
}

// ---------------------------------------------------------------------------
// Kernel 5: normalize and re-layout to [b, p, h*64+d] as f16 for the out GEMM.
// ---------------------------------------------------------------------------
__global__ __launch_bounds__(256) void norm_kernel(
    const float* __restrict__ Ysum, const float* __restrict__ Asum,
    _Float16* __restrict__ Attn) {
  size_t gid = (size_t)blockIdx.x * blockDim.x + threadIdx.x;
  if (gid >= (size_t)BH * SEQ * DIMH) return;
  int d = (int)(gid & 63);
  size_t rem = gid >> 6;
  int p = (int)(rem & 4095);
  int bh = (int)(rem >> 12);
  int b = bh >> 4, h = bh & 15;
  float a = Asum[(size_t)bh * SEQ + p] + 1e-8f;
  Attn[((size_t)(b * SEQ + p)) * INNER + h * DIMH + d] =
      (_Float16)(Ysum[gid] / a);
}

// ---------------------------------------------------------------------------
// Kernel 6: out = attn @ w_out^T + b_out (f32). A tiles (f16, 256x32) are
// DMA'd into LDS by the Tensor Data Mover (wave 0 issues, s_wait_tensorcnt).
// ---------------------------------------------------------------------------
__global__ __launch_bounds__(256) void out_gemm_kernel(
    const _Float16* __restrict__ A, const float* __restrict__ W,
    const float* __restrict__ bias, float* __restrict__ out) {
  __shared__ _Float16 As[256 * 40];
  __shared__ _Float16 Bs[64 * 40];
  const int t = threadIdx.x, w = t >> 5, lane = t & 31;
  const int hi = lane >> 4, nn = lane & 15;
  const int n0 = blockIdx.x * 64, m0 = blockIdx.y * 256;
  const unsigned as_lds = (unsigned)(uintptr_t)(&As[0]);

  v8f acc[2][4];
#pragma unroll
  for (int s = 0; s < 2; ++s)
#pragma unroll
    for (int j = 0; j < 4; ++j) acc[s][j] = v8f_zero();

  for (int k0 = 0; k0 < INNER; k0 += 32) {
    if (w == 0) {                            // TDM: A tile 256x32 halves
      tdm_load_tile_f16(A + (size_t)m0 * INNER + k0, as_lds, 256, 32, INNER);
    }
#pragma unroll
    for (int i = 0; i < 2; ++i) {            // B: 64x32 f32 -> f16
      int idx = t + 256 * i;
      int row = idx >> 3, c = (idx & 7) * 4;
      float4 v = *(const float4*)(W + (size_t)(n0 + row) * INNER + k0 + c);
      _Float16* d = &Bs[row * 40 + c];
      d[0] = (_Float16)v.x; d[1] = (_Float16)v.y;
      d[2] = (_Float16)v.z; d[3] = (_Float16)v.w;
    }
    if (k0 + 32 < INNER)
      __builtin_prefetch(W + (size_t)(n0 + (t & 63)) * INNER + k0 + 32, 0, 0);
    if (w == 0) __builtin_amdgcn_s_wait_tensorcnt(0);
    __syncthreads();
    v16h a0 = frag_load(&As[(w * 32) * 40], 40);
    v16h a1 = frag_load(&As[(w * 32 + 16) * 40], 40);
#pragma unroll
    for (int j = 0; j < 4; ++j) {
      v16h b = frag_load(&Bs[j * 16 * 40], 40);
      acc[0][j] = wmma_f16(a0, b, acc[0][j]);
      acc[1][j] = wmma_f16(a1, b, acc[1][j]);
    }
    __syncthreads();
  }

#pragma unroll
  for (int s = 0; s < 2; ++s) {
#pragma unroll
    for (int j = 0; j < 4; ++j) {
      int nq = n0 + j * 16 + nn;
      float bb = bias[nq];
#pragma unroll
      for (int r = 0; r < 8; ++r) {
        int m = m0 + w * 32 + s * 16 + r + 8 * hi;
        out[(size_t)m * INNER + nq] = acc[s][j][r] + bb;
      }
    }
  }
}

// ---------------------------------------------------------------------------
extern "C" void kernel_launch(void* const* d_in, const int* in_sizes, int n_in,
                              void* d_out, int out_size, void* d_ws,
                              size_t ws_size, hipStream_t stream) {
  (void)in_sizes; (void)n_in; (void)out_size; (void)ws_size;
  const float* X    = (const float*)d_in[0];  // [4,4096,1024]
  const float* Wqkv = (const float*)d_in[1];  // [3072,1024]
  const float* Wout = (const float*)d_in[2];  // [1024,1024]
  const float* Bout = (const float*)d_in[3];  // [1024]
  float* out = (float*)d_out;                 // [4,4096,1024]

  char* ws = (char*)d_ws;
  const size_t szQ0 = (size_t)BH * SEQ * DIMH * sizeof(_Float16);
  const size_t szPy = (size_t)BH * PYRLEN * DIMH * sizeof(_Float16);
  _Float16* Q0 = (_Float16*)ws; ws += szQ0;
  _Float16* K0 = (_Float16*)ws; ws += szQ0;
  _Float16* V0 = (_Float16*)ws; ws += szQ0;
  _Float16* Pq = (_Float16*)ws; ws += szPy;
  _Float16* Pk = (_Float16*)ws; ws += szPy;
  _Float16* Pv = (_Float16*)ws; ws += szPy;
  float* Ysum = (float*)ws; ws += (size_t)BH * SEQ * DIMH * sizeof(float);
  float* Asum = (float*)ws; ws += (size_t)BH * SEQ * sizeof(float);
  _Float16* Attn = (_Float16*)ws; ws += szQ0;

  // 1. QKV projection (WMMA) with head-split f16 epilogue.
  qkv_gemm_kernel<<<dim3(NQKV / 64, (BATCH * SEQ) / 256), 256, 0, stream>>>(
      X, Wqkv, Q0, K0, V0);

  // 2. Build coarsened pyramid, levels 1..7.
  const _Float16 *sq = Q0, *sk = K0, *sv = V0;
  for (int l = 1; l <= 7; ++l) {
    int Ld = SEQ >> l;
    size_t off = (size_t)BH * DIMH * (size_t)(SEQ - (8192 >> l));
    _Float16 *dq = Pq + off, *dk = Pk + off, *dv = Pv + off;
    size_t total = (size_t)BH * Ld * DIMH;
    pyr_kernel<<<(unsigned)((total + 255) / 256), 256, 0, stream>>>(
        sq, sk, sv, dq, dk, dv, Ld);
    sq = dq; sk = dk; sv = dv;
  }

  // 3. Level-0 diagonal causal attention (initializes Ysum/Asum).
  attn_l0_kernel<<<dim3(SEQ / 16, BH), 32, 0, stream>>>(Q0, K0, V0, Ysum, Asum);

  // 4. Levels 1..7, ordered launches (deterministic +=).
  for (int l = 1; l <= 7; ++l)
    attn_off_kernel<<<dim3(128 >> l, BH), 32, 0, stream>>>(Pq, Pk, Pv, Ysum,
                                                           Asum, l);

  // 5. Normalize + relayout.
  norm_kernel<<<(unsigned)(((size_t)BH * SEQ * DIMH + 255) / 256), 256, 0,
                stream>>>(Ysum, Asum, Attn);

  // 6. Output projection (WMMA, TDM-staged A) + bias.
  out_gemm_kernel<<<dim3(INNER / 64, (BATCH * SEQ) / 256), 256, 0, stream>>>(
      Attn, Wout, Bout, out);
}
